// EventMemoryCell_180388627365
// MI455X (gfx1250) — compile-verified, matrix operand order checked by reading, not run on hardware
//
#include <hip/hip_runtime.h>
#include <cstddef>

// ---------------------------------------------------------------------------
// EventMemoryCell forward for MI455X (gfx1250), bf16 WMMA path.
// B=4096, D=H=512, S=16, F=1025 (padded to 1056), 4H=2048, 2H+D=1536.
// Double-buffered GLOBAL_LOAD_ASYNC_TO_LDS_B128 staging (ASYNCcnt) for the
// transB GEMMs, guarded by __has_builtin with sync fallback.
// ---------------------------------------------------------------------------

typedef __attribute__((ext_vector_type(16))) __bf16 v16bf;
typedef __attribute__((ext_vector_type(8)))  float  v8f;

#define TB   256     // threads per block (8 waves, wave32)
#define BM   128
#define BN   128
#define BKK  32
#define LDT  40      // padded LDS row (elements); 80B rows keep b128 alignment

#if defined(__HIP_DEVICE_COMPILE__) && \
    __has_builtin(__builtin_amdgcn_global_load_async_to_lds_b128)
#define HAVE_ASYNC 1
#else
#define HAVE_ASYNC 0
#endif

__device__ __forceinline__ unsigned short f32_to_bf16(float f) {
    unsigned int u = __float_as_uint(f);
    unsigned int r = u + 0x7FFFu + ((u >> 16) & 1u);   // round-to-nearest-even
    return (unsigned short)(r >> 16);
}
__device__ __forceinline__ float sigmoidf(float x) { return 1.0f / (1.0f + expf(-x)); }

union FragAB { v16bf v; unsigned int u[8]; };
union FragC  { v8f   v; float         f[8]; };

// K offset pattern for 16-bit A/B fragments (ISA 7.12.2, 16-bit A 16x32):
// VGPR i<4 : K = 2i + 8*half ; VGPR i>=4 : K = 16 + 2(i-4) + 8*half
__device__ __forceinline__ int frag_koff(int i, int half) {
    return (i < 4) ? (2 * i + 8 * half) : (16 + 2 * (i - 4) + 8 * half);
}

#if HAVE_ASYNC
typedef int v4i_ty __attribute__((vector_size(16)));
typedef __attribute__((address_space(1))) v4i_ty* as1_v4i;
typedef __attribute__((address_space(3))) v4i_ty* as3_v4i;

__device__ __forceinline__ void async_copy_b128(const unsigned short* g,
                                                unsigned short* l) {
    __builtin_amdgcn_global_load_async_to_lds_b128((as1_v4i)g, (as3_v4i)l, 0, 0);
}
#endif

template <int N>
__device__ __forceinline__ void wait_async() {
#if HAVE_ASYNC
#if __has_builtin(__builtin_amdgcn_s_wait_asynccnt)
    __builtin_amdgcn_s_wait_asynccnt(N);
#else
    asm volatile("s_wait_asynccnt %0" :: "i"(N) : "memory");
#endif
#endif
}

// One 32-deep K step: load 4 A frags + 2 B frags from LDS, 8 WMMAs.
__device__ __forceinline__ void mma_tile(const unsigned short* __restrict__ sA,
                                         const unsigned short* __restrict__ sB,
                                         FragC (&acc)[4][2],
                                         int wm, int wn, int l16, int half) {
    FragAB fa[4], fb[2];
#pragma unroll
    for (int s = 0; s < 4; ++s) {
        int row = wm * 64 + s * 16 + l16;
#pragma unroll
        for (int i = 0; i < 8; ++i)
            fa[s].u[i] = *(const unsigned int*)(sA + row * LDT + frag_koff(i, half));
    }
#pragma unroll
    for (int n = 0; n < 2; ++n) {
        int col = wn * 32 + n * 16 + l16;
#pragma unroll
        for (int i = 0; i < 8; ++i)
            fb[n].u[i] = *(const unsigned int*)(sB + col * LDT + frag_koff(i, half));
    }
#pragma unroll
    for (int s = 0; s < 4; ++s)
#pragma unroll
        for (int n = 0; n < 2; ++n)
            acc[s][n].v = __builtin_amdgcn_wmma_f32_16x16x32_bf16(
                false, fa[s].v, false, fb[n].v, (short)0, acc[s][n].v, false, false);
}

// ---------------------------------------------------------------------------
// Generic bf16 GEMM:  C(MxN,f32) [+]= A(MxK,bf16) * B(bf16) [+ bias] [act]
//   TRANSB=true : B stored row-major (N,K)  (computes A * Brm^T)
//   TRANSB=false: B stored row-major (K,N)
// Optional bf16 mirror of C. M%128==0, N%128==0, K%32==0 required.
// ---------------------------------------------------------------------------
template <bool TRANSB>
__global__ __launch_bounds__(TB)
void gemm_bf16_kernel(const unsigned short* __restrict__ A,
                      const unsigned short* __restrict__ Bm,
                      float* __restrict__ C,
                      unsigned short* __restrict__ Cbf,
                      const float* __restrict__ bias,
                      int M, int N, int K, int act, int accum) {
    __shared__ unsigned short sA[2][BM * LDT];
    __shared__ unsigned short sB[2][BN * LDT];

    const int t    = threadIdx.x;
    const int lane = t & 31;
    const int wave = t >> 5;        // 0..7
    const int wm   = wave >> 2;     // 0..1 : 64-row strip
    const int wn   = wave & 3;      // 0..3 : 32-col strip
    const int m0   = blockIdx.y * BM;
    const int n0   = blockIdx.x * BN;
    const int half = lane >> 4;
    const int l16  = lane & 15;

    FragC acc[4][2];
#pragma unroll
    for (int s = 0; s < 4; ++s)
#pragma unroll
        for (int n = 0; n < 2; ++n)
#pragma unroll
            for (int j = 0; j < 8; ++j) acc[s][n].f[j] = 0.0f;

#if HAVE_ASYNC
    if constexpr (TRANSB) {
        // -------- async double-buffered pipeline (ASYNCcnt tracked) --------
        auto issue_tile = [&](int kt, int buf) {
#pragma unroll
            for (int it = 0; it < 2; ++it) {          // A: 2 x b128 per thread
                int idx = t + it * TB;
                int r = idx >> 2, c8 = (idx & 3) * 8;
                async_copy_b128(A + (size_t)(m0 + r) * K + kt * BKK + c8,
                                &sA[buf][r * LDT + c8]);
            }
#pragma unroll
            for (int it = 0; it < 2; ++it) {          // B: 2 x b128 per thread
                int idx = t + it * TB;
                int r = idx >> 2, c8 = (idx & 3) * 8;
                async_copy_b128(Bm + (size_t)(n0 + r) * K + kt * BKK + c8,
                                &sB[buf][r * LDT + c8]);
            }
        };
        const int KT = K / BKK;
        issue_tile(0, 0);
        for (int kt = 0; kt < KT; ++kt) {
            const int cur = kt & 1;
            if (kt + 1 < KT) {
                issue_tile(kt + 1, cur ^ 1);   // prefetch next tile (other buffer)
                wait_async<4>();               // current tile's 4 ops complete
            } else {
                wait_async<0>();
            }
            __syncthreads();
            mma_tile(sA[cur], sB[cur], acc, wm, wn, l16, half);
            __syncthreads();                   // readers done before next overwrite
        }
    } else
#endif
    {
        // -------- synchronous fallback / transpose-on-load path --------
        for (int k0 = 0; k0 < K; k0 += BKK) {
#pragma unroll
            for (int it = 0; it < 2; ++it) {
                int idx = t + it * TB;
                int r = idx >> 2, c8 = (idx & 3) * 8;
                const uint4 val = *(const uint4*)(A + (size_t)(m0 + r) * K + k0 + c8);
                unsigned int* dst = (unsigned int*)(&sA[0][r * LDT + c8]);
                dst[0] = val.x; dst[1] = val.y; dst[2] = val.z; dst[3] = val.w;
            }
            if (TRANSB) {
#pragma unroll
                for (int it = 0; it < 2; ++it) {
                    int idx = t + it * TB;
                    int r = idx >> 2, c8 = (idx & 3) * 8;
                    const uint4 val = *(const uint4*)(Bm + (size_t)(n0 + r) * K + k0 + c8);
                    unsigned int* dst = (unsigned int*)(&sB[0][r * LDT + c8]);
                    dst[0] = val.x; dst[1] = val.y; dst[2] = val.z; dst[3] = val.w;
                }
            } else {
                // global (K,N) -> LDS (N,K) transpose (small GEMMs only)
#pragma unroll
                for (int it = 0; it < 16; ++it) {
                    int lin = it * TB + t;
                    int kk = lin >> 7, nn = lin & 127;
                    sB[0][nn * LDT + kk] = Bm[(size_t)(k0 + kk) * N + n0 + nn];
                }
            }
            __syncthreads();
            mma_tile(sA[0], sB[0], acc, wm, wn, l16, half);
            __syncthreads();
        }
    }

    // ---- epilogue: C/D layout = VGPR j -> M = 8*half + j ; lane -> N = l16 ----
#pragma unroll
    for (int s = 0; s < 4; ++s)
#pragma unroll
        for (int n = 0; n < 2; ++n) {
            int gmb = m0 + wm * 64 + s * 16 + half * 8;
            int gn  = n0 + wn * 32 + n * 16 + l16;
            float bv = bias ? bias[gn] : 0.0f;
#pragma unroll
            for (int j = 0; j < 8; ++j) {
                size_t off = (size_t)(gmb + j) * N + gn;
                float x = acc[s][n].f[j] + bv;
                if (accum) x += C[off];
                if (act == 1) x = tanhf(x);
                C[off] = x;
                if (Cbf) Cbf[off] = f32_to_bf16(x);
            }
        }
}

// ---------------------------------------------------------------------------
// Small helper kernels
// ---------------------------------------------------------------------------
__global__ void cvt_kernel(const float* __restrict__ in, unsigned short* __restrict__ out,
                           size_t n) {
    size_t i = (size_t)blockIdx.x * blockDim.x + threadIdx.x;
    if (i < n) out[i] = f32_to_bf16(in[i]);
}

__global__ void cvt_pad_kernel(const float* __restrict__ in, unsigned short* __restrict__ out,
                               int rows, int cin, int cout) {
    size_t i = (size_t)blockIdx.x * blockDim.x + threadIdx.x;
    size_t n = (size_t)rows * cout;
    if (i >= n) return;
    int r = (int)(i / cout), c = (int)(i % cout);
    out[i] = (c < cin) ? f32_to_bf16(in[(size_t)r * cin + c]) : (unsigned short)0;
}

__global__ void bias_sum_kernel(const float* a, const float* b, float* o, int n) {
    int i = blockIdx.x * blockDim.x + threadIdx.x;
    if (i < n) o[i] = a[i] + b[i];
}

// Per-batch scalars: g = sig(max_s slots[b,s,:].u[b]), r = sig(x.Wr+br), leak = sig(x.Wa+ba)
__global__ __launch_bounds__(TB)
void stage_scalars_kernel(const float* __restrict__ x, const float* __restrict__ slots,
                          const float* __restrict__ u, const float* __restrict__ Wr,
                          const float* __restrict__ br, const float* __restrict__ Wa,
                          const float* __restrict__ ba, float* __restrict__ scal,
                          int D, int S) {
    __shared__ float red[TB];
    __shared__ float sims[16];
    __shared__ float misc[2];
    const int b = blockIdx.x, t = threadIdx.x;
    const float* ub = u + (size_t)b * D;
    const float* xb = x + (size_t)b * D;

    for (int s = 0; s < S; ++s) {
        const float* sl = slots + ((size_t)b * S + s) * D;
        float p = 0.f;
        for (int d = t; d < D; d += TB) p += sl[d] * ub[d];
        red[t] = p; __syncthreads();
        for (int off = TB / 2; off > 0; off >>= 1) {
            if (t < off) red[t] += red[t + off];
            __syncthreads();
        }
        if (t == 0) sims[s] = red[0];
        __syncthreads();
    }
    float pr = 0.f, pa = 0.f;
    for (int d = t; d < D; d += TB) { float xv = xb[d]; pr += xv * Wr[d]; pa += xv * Wa[d]; }
    red[t] = pr; __syncthreads();
    for (int off = TB / 2; off > 0; off >>= 1) { if (t < off) red[t] += red[t + off]; __syncthreads(); }
    if (t == 0) misc[0] = red[0];
    __syncthreads();
    red[t] = pa; __syncthreads();
    for (int off = TB / 2; off > 0; off >>= 1) { if (t < off) red[t] += red[t + off]; __syncthreads(); }
    if (t == 0) misc[1] = red[0];
    __syncthreads();
    if (t == 0) {
        float mx = sims[0];
        for (int s = 1; s < S; ++s) mx = fmaxf(mx, sims[s]);
        scal[b * 3 + 0] = sigmoidf(mx);
        scal[b * 3 + 1] = sigmoidf(misc[0] + br[0]);
        scal[b * 3 + 2] = sigmoidf(misc[1] + ba[0]);
    }
}

// Build step-s LSTM input row (B x FP bf16): [slots_new | cum_new | delta | 0-pad]
__global__ __launch_bounds__(TB)
void build_step_kernel(const float* __restrict__ x, const float* __restrict__ slots,
                       const float* __restrict__ cum, const float* __restrict__ delta,
                       const float* __restrict__ v, const float* __restrict__ scal,
                       unsigned short* __restrict__ out, int D, int S, int FP, int s) {
    const int b = blockIdx.x;
    const float g = scal[b * 3 + 0], r = scal[b * 3 + 1], lk = scal[b * 3 + 2];
    const float* xb = x + (size_t)b * D;
    unsigned short* ob = out + (size_t)b * FP;
    for (int idx = threadIdx.x; idx < FP; idx += TB) {
        float val;
        if (idx < D) {
            int d = idx;
            float srow;
            if (s < S - 1) srow = slots[((size_t)b * S + s + 1) * D + d];
            else srow = g * (r * slots[(size_t)b * S * D + d] + (1.f - r) * v[(size_t)b * D + d]);
            val = srow + lk * xb[d];
        } else if (idx < 2 * D) {
            int d = idx - D;
            float crow = (s < S - 1) ? cum[((size_t)b * S + s + 1) * D + d] : 0.f;
            val = crow + 2.f * xb[d];          // cum gets x_t added twice in the reference
        } else if (idx == 2 * D) {
            val = (s < S - 1) ? (delta[(size_t)b * S + s + 1] + 1.f) : 0.f;
        } else {
            val = 0.f;                          // K padding
        }
        ob[idx] = f32_to_bf16(val);
    }
}

// LSTM gate pointwise (gate order i,f,g,o)
__global__ void lstm_point_kernel(const float* __restrict__ gates, float* __restrict__ c,
                                  float* __restrict__ h, unsigned short* __restrict__ hbf,
                                  int B, int H, int first) {
    int idx = blockIdx.x * blockDim.x + threadIdx.x;
    if (idx >= B * H) return;
    int b = idx / H, hh = idx - b * H;
    const float* gb = gates + (size_t)b * 4 * H;
    float gi = gb[hh], gf = gb[H + hh], gg = gb[2 * H + hh], go = gb[3 * H + hh];
    float cp = first ? 0.f : c[idx];
    float cn = sigmoidf(gf) * cp + sigmoidf(gi) * tanhf(gg);
    float hn = sigmoidf(go) * tanhf(cn);
    c[idx] = cn; h[idx] = hn; hbf[idx] = f32_to_bf16(hn);
}

// concat [mem_out | h_prev | x_t] -> bf16 (B x (2H+D))
__global__ void concat_kernel(const float* __restrict__ h, const float* __restrict__ hprev,
                              const float* __restrict__ x, unsigned short* __restrict__ out,
                              int B, int H, int D) {
    int tot = 2 * H + D;
    size_t i = (size_t)blockIdx.x * blockDim.x + threadIdx.x;
    if (i >= (size_t)B * tot) return;
    int b = (int)(i / tot), j = (int)(i % tot);
    float val;
    if (j < H) val = h[(size_t)b * H + j];
    else if (j < 2 * H) val = hprev[(size_t)b * H + (j - H)];
    else val = x[(size_t)b * D + (j - 2 * H)];
    out[i] = f32_to_bf16(val);
}

// ---------------------------------------------------------------------------
// Host orchestration
// ---------------------------------------------------------------------------
static inline void* wsa(char*& p, size_t bytes) {
    void* r = (void*)p;
    p += (bytes + 255) & ~(size_t)255;
    return r;
}

extern "C" void kernel_launch(void* const* d_in, const int* in_sizes, int n_in,
                              void* d_out, int out_size, void* d_ws, size_t ws_size,
                              hipStream_t stream) {
    (void)in_sizes; (void)n_in; (void)out_size; (void)ws_size;
    const int B = 4096, D = 512, H = 512, S = 16;
    const int F = 2 * D + 1;          // 1025
    const int FP = 1056;              // padded K (multiple of 32)
    const int G4 = 4 * H;             // 2048
    const int C2 = 2 * H + D;         // 1536

    const float* x_t    = (const float*)d_in[0];
    const float* h_prev = (const float*)d_in[1];
    const float* slots  = (const float*)d_in[2];
    const float* cum    = (const float*)d_in[3];
    const float* delta  = (const float*)d_in[4];
    const float* Wk     = (const float*)d_in[5];
    const float* Wq     = (const float*)d_in[6];
    const float* Wv     = (const float*)d_in[7];
    const float* bv     = (const float*)d_in[8];
    const float* Wr     = (const float*)d_in[9];
    const float* br     = (const float*)d_in[10];
    const float* Wa     = (const float*)d_in[11];
    const float* ba     = (const float*)d_in[12];
    const float* W_ih   = (const float*)d_in[13];
    const float* W_hh   = (const float*)d_in[14];
    const float* b_ih   = (const float*)d_in[15];
    const float* b_hh   = (const float*)d_in[16];
    const float* Wo     = (const float*)d_in[17];
    const float* bo     = (const float*)d_in[18];
    float* out          = (float*)d_out;

    char* p = (char*)d_ws;
    unsigned short* xb    = (unsigned short*)wsa(p, (size_t)B * D * 2);
    unsigned short* Wq_b  = (unsigned short*)wsa(p, (size_t)H * D * 2);
    unsigned short* Wk_b  = (unsigned short*)wsa(p, (size_t)H * D * 2);
    unsigned short* Wv_b  = (unsigned short*)wsa(p, (size_t)D * D * 2);
    unsigned short* Wih_b = (unsigned short*)wsa(p, (size_t)G4 * FP * 2);
    unsigned short* Whh_b = (unsigned short*)wsa(p, (size_t)G4 * H * 2);
    unsigned short* Wo_b  = (unsigned short*)wsa(p, (size_t)H * C2 * 2);
    float*          bsum  = (float*)wsa(p, (size_t)G4 * 4);
    float*          ctmp  = (float*)wsa(p, (size_t)B * H * 4);
    unsigned short* qb    = (unsigned short*)wsa(p, (size_t)B * H * 2);
    float*          u     = (float*)wsa(p, (size_t)B * D * 4);
    float*          v     = (float*)wsa(p, (size_t)B * D * 4);
    float*          scal  = (float*)wsa(p, (size_t)B * 3 * 4);
    unsigned short* ins   = (unsigned short*)wsa(p, (size_t)B * FP * 2);
    float*          gates = (float*)wsa(p, (size_t)B * G4 * 4);
    float*          cbuf  = (float*)wsa(p, (size_t)B * H * 4);
    float*          hbuf  = (float*)wsa(p, (size_t)B * H * 4);
    unsigned short* hbf   = (unsigned short*)wsa(p, (size_t)B * H * 2);
    unsigned short* cat   = (unsigned short*)wsa(p, (size_t)B * C2 * 2);

    auto nb = [](size_t n) { return (unsigned)((n + TB - 1) / TB); };

    // --- conversions ---
    cvt_kernel<<<nb((size_t)B * D), TB, 0, stream>>>(x_t, xb, (size_t)B * D);
    cvt_kernel<<<nb((size_t)H * D), TB, 0, stream>>>(Wq, Wq_b, (size_t)H * D);
    cvt_kernel<<<nb((size_t)H * D), TB, 0, stream>>>(Wk, Wk_b, (size_t)H * D);
    cvt_kernel<<<nb((size_t)D * D), TB, 0, stream>>>(Wv, Wv_b, (size_t)D * D);
    cvt_kernel<<<nb((size_t)G4 * H), TB, 0, stream>>>(W_hh, Whh_b, (size_t)G4 * H);
    cvt_kernel<<<nb((size_t)H * C2), TB, 0, stream>>>(Wo, Wo_b, (size_t)H * C2);
    cvt_pad_kernel<<<nb((size_t)G4 * FP), TB, 0, stream>>>(W_ih, Wih_b, G4, F, FP);
    bias_sum_kernel<<<nb(G4), TB, 0, stream>>>(b_ih, b_hh, bsum, G4);

    // --- small GEMMs: q = x Wq^T (bf16 mirror), u = q Wk, v = x Wv^T + bv ---
    {
        dim3 g(H / BN, B / BM);
        gemm_bf16_kernel<true><<<g, TB, 0, stream>>>(xb, Wq_b, ctmp, qb, nullptr,
                                                     B, H, D, 0, 0);
    }
    {
        dim3 g(D / BN, B / BM);
        gemm_bf16_kernel<false><<<g, TB, 0, stream>>>(qb, Wk_b, u, nullptr, nullptr,
                                                      B, D, H, 0, 0);
        gemm_bf16_kernel<true><<<g, TB, 0, stream>>>(xb, Wv_b, v, nullptr, bv,
                                                     B, D, D, 0, 0);
    }

    // --- per-batch gating scalars ---
    stage_scalars_kernel<<<B, TB, 0, stream>>>(x_t, slots, u, Wr, br, Wa, ba, scal, D, S);

    // --- 16-step LSTM over slots ---
    dim3 gG(G4 / BN, B / BM);
    for (int s = 0; s < S; ++s) {
        build_step_kernel<<<B, TB, 0, stream>>>(x_t, slots, cum, delta, v, scal,
                                                ins, D, S, FP, s);
        gemm_bf16_kernel<true><<<gG, TB, 0, stream>>>(ins, Wih_b, gates, nullptr, bsum,
                                                      B, G4, FP, 0, 0);
        if (s > 0)
            gemm_bf16_kernel<true><<<gG, TB, 0, stream>>>(hbf, Whh_b, gates, nullptr, nullptr,
                                                          B, G4, H, 0, 1);
        lstm_point_kernel<<<nb((size_t)B * H), TB, 0, stream>>>(gates, cbuf, hbuf, hbf,
                                                                B, H, (s == 0) ? 1 : 0);
    }

    // --- output projection: tanh(concat(mem_out, h_prev, x_t) Wo^T + bo) ---
    concat_kernel<<<nb((size_t)B * C2), TB, 0, stream>>>(hbuf, h_prev, x_t, cat, B, H, D);
    {
        dim3 g(H / BN, B / BM);
        gemm_bf16_kernel<true><<<g, TB, 0, stream>>>(cat, Wo_b, out, nullptr, bo,
                                                     B, H, C2, 1, 0);
    }
}
